// HeteroDecoder_72146860638641
// MI455X (gfx1250) — matrix operand order repaired and checked by reading, not compiled, not run on hardware
//
#include <hip/hip_runtime.h>

#define HID 128
#define WPC 128   // workgroups per combo

typedef __attribute__((ext_vector_type(16))) __bf16 v16bf;
typedef __attribute__((ext_vector_type(2)))  __bf16 v2bf;
typedef __attribute__((ext_vector_type(8)))  float  v8f;

union V16 {
  v16bf v;
  unsigned int u[8];
};

// packed f32x2 -> bf16x2 (RNE) using the hardware convert when available
__device__ __forceinline__ unsigned int f2bf2(float lo, float hi) {
#if __has_builtin(__builtin_amdgcn_cvt_pk_bf16_f32)
  v2bf r = __builtin_amdgcn_cvt_pk_bf16_f32(lo, hi);
  return __builtin_bit_cast(unsigned int, r);
#else
  v2bf r = { (__bf16)lo, (__bf16)hi };
  return __builtin_bit_cast(unsigned int, r);
#endif
}
__device__ __forceinline__ unsigned short f2bf(float f) {
  return __builtin_bit_cast(unsigned short, (__bf16)f);
}

// combos-with-replacement index; 6 == "no combo matches" (gs > ge)
__device__ __forceinline__ int combo_of(int vs, int ve) {
  int gs = vs / 3, ge = ve / 3;             // GROUPS are [0..2],[3..5],[6..8]
  if (gs > ge) return 6;
  return gs * 3 + ge - ((gs * (gs + 1)) >> 1);
}

// ---------------------------------------------------------------- kernel 0
__global__ void init_kernel(float* out, int* ws, int E) {
  int i = blockIdx.x * blockDim.x + threadIdx.x;
  if (i < E) out[i] = 0.f;
  if (i < 32) ws[i] = 0;
}

// ---------------------------------------------------------------- kernel 1
__global__ void count_kernel(const int* __restrict__ eidx,
                             const int* __restrict__ vol,
                             int* __restrict__ ws, int E) {
  __shared__ int lc[8];
  int tid = threadIdx.x;
  if (tid < 8) lc[tid] = 0;
  __syncthreads();
  int i = blockIdx.x * blockDim.x + tid;
  if (i < E) {
    int c = combo_of(vol[eidx[i]], vol[eidx[E + i]]);
    if (c < 6) atomicAdd(&lc[c], 1);
  }
  __syncthreads();
  if (tid < 6 && lc[tid]) atomicAdd(&ws[tid], lc[tid]);
}

// ---------------------------------------------------------------- kernel 2
__global__ void scan_kernel(int* ws) {
  int o = 0;
  for (int c = 0; c < 6; ++c) {
    ws[8 + c]  = o;   // segment base (read by gemm)
    ws[16 + c] = o;   // running fill position
    o += ws[c];
  }
}

// ---------------------------------------------------------------- kernel 3
__global__ void fill_kernel(const int* __restrict__ eidx,
                            const int* __restrict__ vol,
                            int* __restrict__ ws, int* __restrict__ lst, int E) {
  __shared__ int lc[6], lb[6];
  int tid = threadIdx.x;
  if (tid < 6) lc[tid] = 0;
  __syncthreads();
  int i = blockIdx.x * blockDim.x + tid;
  int c = 6, lp = 0;
  if (i < E) {
    c = combo_of(vol[eidx[i]], vol[eidx[E + i]]);
    if (c < 6) lp = atomicAdd(&lc[c], 1);
  }
  __syncthreads();
  if (tid < 6) lb[tid] = lc[tid] ? atomicAdd(&ws[16 + tid], lc[tid]) : 0;
  __syncthreads();
  if (c < 6) lst[lb[c] + lp] = i;
}

// ---------------------------------------------------------------- kernel 4
// LDS weight layout is [lane][tile]: lane L keeps its 32-byte slice of every
// WMMA B tile contiguously (16B pad per lane de-stripes banks), so B-operand
// ds_load_b128s are base+imm. Per-wave h tiles round-trip D-layout -> A-layout.
#define W1_STRIDE 1544   // 96 tiles * 16 ushorts + 8 pad
#define W2_STRIDE 520    // 32 tiles * 16 ushorts + 8 pad
__global__ __launch_bounds__(256) void hetero_gemm(
    const float* __restrict__ x,  const int*   __restrict__ eidx,
    const float* __restrict__ ef,
    const float* __restrict__ W1, const float* __restrict__ b1,
    const float* __restrict__ g1, const float* __restrict__ be1,
    const float* __restrict__ W2, const float* __restrict__ b2,
    const float* __restrict__ g2, const float* __restrict__ be2,
    const float* __restrict__ W3, const float* __restrict__ b3,
    const int*   __restrict__ ws, const int*  __restrict__ lst,
    float* __restrict__ out, int E)
{
  __shared__ __align__(16) unsigned short W1s[32 * W1_STRIDE]; // 98816 B
  __shared__ __align__(16) unsigned short W2s[32 * W2_STRIDE]; // 33280 B
  __shared__ __align__(16) unsigned short Hs [8 * 16 * HID];   // 32768 B

  int c     = blockIdx.x / WPC;
  int wslot = blockIdx.x % WPC;
  int tid   = threadIdx.x;

  // ---- stage W1 (384x128) and W2 (128x128): pairs along k -> u32 stores ----
  const float* W1g = W1 + (size_t)c * (3 * HID * HID);
  for (int p = tid; p < 3 * HID * HID / 2; p += 256) {
    int n = p & 127;
    int k = (p >> 7) * 2;                       // even k; k,k+1 share lane/tile
    float lo  = W1g[k * HID + n];
    float hi2 = W1g[k * HID + HID + n];
    int ln   = (n & 15) + ((k & 8) ? 16 : 0);
    int tile = (k >> 5) * 8 + (n >> 4);
    *reinterpret_cast<unsigned int*>(&W1s[ln * W1_STRIDE + tile * 16 + (k & 7) + ((k & 16) >> 1)]) =
        f2bf2(lo, hi2);
  }
  const float* W2g = W2 + (size_t)c * (HID * HID);
  for (int p = tid; p < HID * HID / 2; p += 256) {
    int n = p & 127;
    int k = (p >> 7) * 2;
    float lo  = W2g[k * HID + n];
    float hi2 = W2g[k * HID + HID + n];
    int ln   = (n & 15) + ((k & 8) ? 16 : 0);
    int tile = (k >> 5) * 8 + (n >> 4);
    *reinterpret_cast<unsigned int*>(&W2s[ln * W2_STRIDE + tile * 16 + (k & 7) + ((k & 16) >> 1)]) =
        f2bf2(lo, hi2);
  }
  __syncthreads();

  int segBase = ws[8 + c];
  int segCnt  = ws[c];
  int nTiles  = (segCnt + 15) >> 4;

  int wave = tid >> 5, lane = tid & 31;
  int m    = lane & 15;           // row handled for A-gather
  int hi   = lane >> 4;           // K-half for A, M-half for C/D

  const unsigned short* w1lane = &W1s[lane * W1_STRIDE];
  const unsigned short* w2lane = &W2s[lane * W2_STRIDE];
  unsigned short* hbuf = &Hs[wave * 16 * HID];
  unsigned short*       hst = hbuf + (hi ? 8 * HID : 0) + m;   // LN store base
  const unsigned short* hld = hbuf + m * HID + (hi ? 8 : 0);   // A2 load base

  // per-lane column constants (column = nb*16 + m), hoisted out of tile loop
  float b1v[8], g1v[8], e1v[8], b2v[8], g2v[8], e2v[8], w3v[8];
#pragma unroll
  for (int nb = 0; nb < 8; ++nb) {
    int n = nb * 16 + m;
    b1v[nb] = b1 [c * HID + n];
    g1v[nb] = g1 [c * HID + n];
    e1v[nb] = be1[c * HID + n];
    b2v[nb] = b2 [c * HID + n];
    g2v[nb] = g2 [c * HID + n];
    e2v[nb] = be2[c * HID + n];
    w3v[nb] = W3 [c * HID + n];
  }
  float b3c = b3[c];

  for (int t = wslot * 8 + wave; t < nTiles; t += WPC * 8) {
    int  row = t * 16 + m;
    bool rv  = row < segCnt;
    int  eid = lst[segBase + (rv ? row : segCnt - 1)];
    int  st  = eidx[eid];
    int  en  = eidx[E + eid];
    const float* s0 = x  + (size_t)st  * HID;
    const float* s1 = x  + (size_t)en  * HID;
    const float* s2 = ef + (size_t)eid * HID;

    // ---- layer 1: kb-outer (1 A block live), nb-inner over accumulators ----
    v8f acc[8];
#pragma unroll
    for (int nb = 0; nb < 8; ++nb) {
      float bv = b1v[nb];
#pragma unroll
      for (int r = 0; r < 8; ++r) acc[nb][r] = bv;
    }
#pragma unroll
    for (int kb = 0; kb < 12; ++kb) {
      const float* src = (kb < 4) ? s0 : ((kb < 8) ? s1 : s2);
      int kin = (kb & 3) * 32;
      int a0  = kin + (hi ? 8 : 0);
      int a1  = kin + (hi ? 24 : 16);
      float4 f0 = *reinterpret_cast<const float4*>(src + a0);
      float4 f1 = *reinterpret_cast<const float4*>(src + a0 + 4);
      float4 f2 = *reinterpret_cast<const float4*>(src + a1);
      float4 f3 = *reinterpret_cast<const float4*>(src + a1 + 4);
      V16 a;
      a.u[0] = f2bf2(f0.x, f0.y); a.u[1] = f2bf2(f0.z, f0.w);
      a.u[2] = f2bf2(f1.x, f1.y); a.u[3] = f2bf2(f1.z, f1.w);
      a.u[4] = f2bf2(f2.x, f2.y); a.u[5] = f2bf2(f2.z, f2.w);
      a.u[6] = f2bf2(f3.x, f3.y); a.u[7] = f2bf2(f3.z, f3.w);
#pragma unroll
      for (int nb = 0; nb < 8; ++nb) {
        V16 B;
        const uint4* bp =
            reinterpret_cast<const uint4*>(w1lane + (kb * 8 + nb) * 16);
        uint4 q0 = bp[0], q1 = bp[1];
        B.u[0] = q0.x; B.u[1] = q0.y; B.u[2] = q0.z; B.u[3] = q0.w;
        B.u[4] = q1.x; B.u[5] = q1.y; B.u[6] = q1.z; B.u[7] = q1.w;
        acc[nb] = __builtin_amdgcn_wmma_f32_16x16x32_bf16(
            false, a.v, false, B.v, (short)0, acc[nb], false, false);
      }
    }

    // ---- LayerNorm1 + ReLU (rows split across half-waves) -> LDS bf16 ----
    {
      float sum[8], sq[8];
#pragma unroll
      for (int r = 0; r < 8; ++r) { sum[r] = 0.f; sq[r] = 0.f; }
#pragma unroll
      for (int nb = 0; nb < 8; ++nb)
#pragma unroll
        for (int r = 0; r < 8; ++r) { float v = acc[nb][r]; sum[r] += v; sq[r] += v * v; }
#pragma unroll
      for (int msk = 1; msk <= 8; msk <<= 1)
#pragma unroll
        for (int r = 0; r < 8; ++r) {
          sum[r] += __shfl_xor(sum[r], msk);
          sq[r]  += __shfl_xor(sq[r],  msk);
        }
#pragma unroll
      for (int r = 0; r < 8; ++r) {
        float mean = sum[r] * (1.f / HID);
        float var  = sq[r] * (1.f / HID) - mean * mean;
        float rstd = rsqrtf(var + 1e-5f);
#pragma unroll
        for (int nb = 0; nb < 8; ++nb) {
          float hn = (acc[nb][r] - mean) * rstd * g1v[nb] + e1v[nb];
          hst[r * HID + nb * 16] = f2bf(fmaxf(hn, 0.f));
        }
      }
    }

    // ---- layer 2: kb-outer, A from per-wave LDS h tile ----
    v8f acc2[8];
#pragma unroll
    for (int nb = 0; nb < 8; ++nb) {
      float bv = b2v[nb];
#pragma unroll
      for (int r = 0; r < 8; ++r) acc2[nb][r] = bv;
    }
#pragma unroll
    for (int kb = 0; kb < 4; ++kb) {
      uint4 q0 = *reinterpret_cast<const uint4*>(hld + kb * 32);
      uint4 q1 = *reinterpret_cast<const uint4*>(hld + kb * 32 + 16);
      V16 a;
      a.u[0] = q0.x; a.u[1] = q0.y; a.u[2] = q0.z; a.u[3] = q0.w;
      a.u[4] = q1.x; a.u[5] = q1.y; a.u[6] = q1.z; a.u[7] = q1.w;
#pragma unroll
      for (int nb = 0; nb < 8; ++nb) {
        V16 B;
        const uint4* bp =
            reinterpret_cast<const uint4*>(w2lane + (kb * 8 + nb) * 16);
        uint4 p0 = bp[0], p1 = bp[1];
        B.u[0] = p0.x; B.u[1] = p0.y; B.u[2] = p0.z; B.u[3] = p0.w;
        B.u[4] = p1.x; B.u[5] = p1.y; B.u[6] = p1.z; B.u[7] = p1.w;
        acc2[nb] = __builtin_amdgcn_wmma_f32_16x16x32_bf16(
            false, a.v, false, B.v, (short)0, acc2[nb], false, false);
      }
    }

    // ---- LayerNorm2 + ReLU fused with W3 dot product ----
    float p[8];
    {
      float sum[8], sq[8];
#pragma unroll
      for (int r = 0; r < 8; ++r) { sum[r] = 0.f; sq[r] = 0.f; p[r] = 0.f; }
#pragma unroll
      for (int nb = 0; nb < 8; ++nb)
#pragma unroll
        for (int r = 0; r < 8; ++r) { float v = acc2[nb][r]; sum[r] += v; sq[r] += v * v; }
#pragma unroll
      for (int msk = 1; msk <= 8; msk <<= 1)
#pragma unroll
        for (int r = 0; r < 8; ++r) {
          sum[r] += __shfl_xor(sum[r], msk);
          sq[r]  += __shfl_xor(sq[r],  msk);
        }
#pragma unroll
      for (int r = 0; r < 8; ++r) {
        float mean = sum[r] * (1.f / HID);
        float var  = sq[r] * (1.f / HID) - mean * mean;
        float rstd = rsqrtf(var + 1e-5f);
#pragma unroll
        for (int nb = 0; nb < 8; ++nb) {
          float hn = (acc2[nb][r] - mean) * rstd * g2v[nb] + e2v[nb];
          p[r] += fmaxf(hn, 0.f) * w3v[nb];
        }
      }
#pragma unroll
      for (int msk = 1; msk <= 8; msk <<= 1)
#pragma unroll
        for (int r = 0; r < 8; ++r) p[r] += __shfl_xor(p[r], msk);
    }

    // lane r holds logit of row r (low half), lane 24+r holds row 8+r and
    // also owns that row's edge id (m == 8+r) from the gather phase.
#pragma unroll
    for (int r = 0; r < 8; ++r) {
      if ((lane == r) || (lane == 24 + r)) {
        if (rv) out[eid] = p[r] + b3c;
      }
    }
  }
}

// ---------------------------------------------------------------- launch
extern "C" void kernel_launch(void* const* d_in, const int* in_sizes, int n_in,
                              void* d_out, int out_size, void* d_ws, size_t ws_size,
                              hipStream_t stream) {
  (void)n_in; (void)out_size; (void)ws_size;
  const float* x    = (const float*)d_in[0];
  const int*   eidx = (const int*)  d_in[1];
  const float* ef   = (const float*)d_in[2];
  const int*   vol  = (const int*)  d_in[3];
  const float* W1   = (const float*)d_in[4];
  const float* b1   = (const float*)d_in[5];
  const float* g1   = (const float*)d_in[6];
  const float* be1  = (const float*)d_in[7];
  const float* W2   = (const float*)d_in[8];
  const float* b2   = (const float*)d_in[9];
  const float* g2   = (const float*)d_in[10];
  const float* be2  = (const float*)d_in[11];
  const float* W3   = (const float*)d_in[12];
  const float* b3   = (const float*)d_in[13];
  float* out = (float*)d_out;
  int E = in_sizes[1] / 2;

  int* ws  = (int*)d_ws;        // [0..5]=cnt [8..13]=base [16..21]=pos
  int* lst = ws + 32;           // E edge ids, segmented per combo

  int nb = (E + 255) / 256;
  init_kernel <<<nb, 256, 0, stream>>>(out, ws, E);
  count_kernel<<<nb, 256, 0, stream>>>(eidx, vol, ws, E);
  scan_kernel <<<1, 1, 0, stream>>>(ws);
  fill_kernel <<<nb, 256, 0, stream>>>(eidx, vol, ws, lst, E);
  hetero_gemm <<<6 * WPC, 256, 0, stream>>>(x, eidx, ef,
                                            W1, b1, g1, be1,
                                            W2, b2, g2, be2,
                                            W3, b3, ws, lst, out, E);
}
